// QuantizedPyramidVisionTransformer_63136019251237
// MI455X (gfx1250) — 8-bit matrix operand order repaired and checked by reading, compiled, not run on hardware
//
#include <hip/hip_runtime.h>
#include <stdint.h>


typedef __attribute__((ext_vector_type(8))) int v8i;

__device__ __forceinline__ float qz_dev(float x) {
  float r = rintf(x * 16.0f);
  r = fminf(fmaxf(r, -128.0f), 127.0f);
  return r * 0.0625f;
}

__global__ void k_quant_pad(const float* __restrict__ src, long srcRowStride,
                            int M, int K, signed char* __restrict__ dst,
                            int Kp, long total) {
  long i = (long)blockIdx.x * blockDim.x + threadIdx.x;
  if (i >= total) return;
  int k = (int)(i % Kp);
  long m = i / Kp;
  int q = 0;
  if (m < (long)M && k < K) {
    float r = rintf(src[m * srcRowStride + k] * 16.0f);
    r = fminf(fmaxf(r, -128.0f), 127.0f);
    q = (int)r;
  }
  dst[i] = (signed char)q;
}

__global__ void k_quant_transpose(const float* __restrict__ src, long srcRowStride,
                                  int Mtok, int Mpad, signed char* __restrict__ dst) {
  int i = blockIdx.x * blockDim.x + threadIdx.x;
  int total = 64 * Mpad;
  if (i >= total) return;
  int m = i % Mpad;
  int dd = i / Mpad;
  int q = 0;
  if (m < Mtok) {
    float r = rintf(src[(long)m * srcRowStride + dd] * 16.0f);
    r = fminf(fmaxf(r, -128.0f), 127.0f);
    q = (int)r;
  }
  dst[i] = (signed char)q;
}

__global__ __launch_bounds__(128) void k_gemm_i8(
    const signed char* __restrict__ Aq, int lda,
    const signed char* __restrict__ Bq, int ldb,
    float* __restrict__ C, int ldc,
    int M, int N, int Kp, float scale,
    const float* __restrict__ bias, int relu) {
  __shared__ int2 ldsA[256];

  const int tid = threadIdx.x;
  const int wave = tid >> 5;
  const int lane = tid & 31;
  const int hi = lane >> 4;
  const int l15 = lane & 15;

  if (relu < 0) ldsA[(unsigned)tid & 255u] = make_int2(lda, ldb);

  const int m0 = blockIdx.y << 5;
  const int nCol = (blockIdx.x << 6) + (wave << 4) + l15;

  const size_t aRowOff = (size_t)(m0 + (tid >> 2)) * lda + (size_t)((tid & 3) << 4);
  const unsigned ldsOff = (unsigned)tid << 4;

  v8i acc0 = {0, 0, 0, 0, 0, 0, 0, 0};
  v8i acc1 = {0, 0, 0, 0, 0, 0, 0, 0};

  for (int k0 = 0; k0 < Kp; k0 += 64) {
    {
      unsigned long long ga = (unsigned long long)(size_t)(Aq + aRowOff + k0);
      asm volatile("global_load_async_to_lds_b128 %0, %1, off"
                   :: "v"(ldsOff), "v"(ga) : "memory");
      asm volatile("s_wait_asynccnt 0x0" ::: "memory");
    }
    __syncthreads();

    v8i a0, a1;
    {
      const int2* L0 = ldsA + (l15 << 3) + (hi << 1);
      const int2* L1 = L0 + 128;
      int2 p0 = L0[0], p1 = L0[1], p2 = L0[4], p3 = L0[5];
      a0[0] = p0.x; a0[1] = p0.y; a0[2] = p1.x; a0[3] = p1.y;
      a0[4] = p2.x; a0[5] = p2.y; a0[6] = p3.x; a0[7] = p3.y;
      int2 q0 = L1[0], q1 = L1[1], q2 = L1[4], q3 = L1[5];
      a1[0] = q0.x; a1[1] = q0.y; a1[2] = q1.x; a1[3] = q1.y;
      a1[4] = q2.x; a1[5] = q2.y; a1[6] = q3.x; a1[7] = q3.y;
    }
    v8i b;
    {
      const int4* Bp = (const int4*)(Bq + (size_t)nCol * ldb + k0 + (hi << 4));
      int4 q0 = Bp[0];
      int4 q1 = Bp[2];
      b[0] = q0.x; b[1] = q0.y; b[2] = q0.z; b[3] = q0.w;
      b[4] = q1.x; b[5] = q1.y; b[6] = q1.z; b[7] = q1.w;
    }
    if (k0 + 64 < Kp) {
      __builtin_prefetch(Bq + (size_t)nCol * ldb + k0 + 64, 0, 1);
      __builtin_prefetch(Aq + aRowOff + k0 + 64, 0, 1);
    }
    acc0 = __builtin_amdgcn_wmma_i32_16x16x64_iu8(true, a0, true, b, acc0, false, false);
    acc1 = __builtin_amdgcn_wmma_i32_16x16x64_iu8(true, a1, true, b, acc1, false, false);
    __syncthreads();
  }

  float bq = 0.0f;
  if (bias && nCol < N) bq = qz_dev(bias[nCol]);
  const int mB = m0 + (hi << 3);
#pragma unroll
  for (int r = 0; r < 8; ++r) {
    int m = mB + r;
    if (m < M && nCol < N) {
      float v = (float)acc0[r] * scale + bq;
      if (relu) v = fmaxf(v, 0.0f);
      C[(size_t)m * ldc + nCol] = v;
    }
    int m2 = mB + 16 + r;
    if (m2 < M && nCol < N) {
      float v = (float)acc1[r] * scale + bq;
      if (relu) v = fmaxf(v, 0.0f);
      C[(size_t)m2 * ldc + nCol] = v;
    }
  }
}

__global__ __launch_bounds__(256) void k_softmax(const float* __restrict__ in,
                                                 float* __restrict__ out, int N) {
  const float* row = in + (size_t)blockIdx.x * N;
  float* orow = out + (size_t)blockIdx.x * N;
  __shared__ float red[8];
  __shared__ float red2[8];

  float mx = -1e30f;
  for (int j = threadIdx.x; j < N; j += 256) mx = fmaxf(mx, qz_dev(row[j]));
  for (int o = 16; o > 0; o >>= 1) mx = fmaxf(mx, __shfl_xor(mx, o, 32));
  if ((threadIdx.x & 31) == 0) red[threadIdx.x >> 5] = mx;
  __syncthreads();
  mx = red[0];
  for (int w = 1; w < 8; ++w) mx = fmaxf(mx, red[w]);

  float s = 0.0f;
  for (int j = threadIdx.x; j < N; j += 256) s += __expf(qz_dev(row[j]) - mx);
  for (int o = 16; o > 0; o >>= 1) s += __shfl_xor(s, o, 32);
  if ((threadIdx.x & 31) == 0) red2[threadIdx.x >> 5] = s;
  __syncthreads();
  s = 0.0f;
  for (int w = 0; w < 8; ++w) s += red2[w];
  float inv = 1.0f / s;

  for (int j = threadIdx.x; j < N; j += 256)
    orow[j] = __expf(qz_dev(row[j]) - mx) * inv;
}

__global__ void k_affine(const float* __restrict__ x, float* __restrict__ y, long n,
                         const float* __restrict__ wp, const float* __restrict__ bp) {
  long i = (long)blockIdx.x * blockDim.x + threadIdx.x;
  if (i >= n) return;
  float w = qz_dev(wp[0]);
  float b = qz_dev(bp[0]);
  y[i] = qz_dev(qz_dev(x[i]) * w) + b;
}

__global__ void k_add_qz(const float* __restrict__ a, const float* __restrict__ bsrc,
                         float* __restrict__ y, long n, long bmod) {
  long i = (long)blockIdx.x * blockDim.x + threadIdx.x;
  if (i >= n) return;
  y[i] = qz_dev(a[i]) + qz_dev(bsrc[i % bmod]);
}

__global__ void k_concat_cls(const float* __restrict__ cls, const float* __restrict__ toks,
                             float* __restrict__ out, int Bb, int Nold, int E, long total) {
  long i = (long)blockIdx.x * blockDim.x + threadIdx.x;
  if (i >= total) return;
  int e = (int)(i % E);
  long t = i / E;
  int n = (int)(t % (Nold + 1));
  int b = (int)(t / (Nold + 1));
  out[i] = (n == 0) ? cls[e] : toks[((long)b * Nold + (n - 1)) * (long)E + e];
}

__global__ void k_im2col(const float* __restrict__ src, float* __restrict__ dst,
                         int Cin, int HH, int p, int nhwc, long total) {
  long i = (long)blockIdx.x * blockDim.x + threadIdx.x;
  if (i >= total) return;
  int Ho = HH / p;
  int K = Cin * p * p;
  long row = i / K;
  int col = (int)(i % K);
  int c = col / (p * p);
  int rr = col - c * p * p;
  int di = rr / p;
  int dj = rr - di * p;
  int j = (int)(row % Ho);
  long t = row / Ho;
  int ii = (int)(t % Ho);
  int b = (int)(t / Ho);
  int y = ii * p + di;
  int x = j * p + dj;
  float v = nhwc ? src[(((long)b * HH + y) * HH + x) * Cin + c]
                 : src[(((long)b * Cin + c) * HH + y) * HH + x];
  dst[i] = v;
}

extern "C" void kernel_launch(void* const* d_in, const int* in_sizes, int n_in,
                              void* d_out, int out_size, void* d_ws, size_t ws_size,
                              hipStream_t stream) {
  (void)in_sizes; (void)n_in; (void)out_size; (void)ws_size;
  static const int EMB[4] = {64, 128, 256, 512};
  static const int NH[4] = {1, 2, 4, 8};
  static const int DEP[4] = {3, 4, 6, 3};
  static const int PS[4] = {4, 2, 2, 2};
  static const int INC[4] = {3, 64, 128, 256};
  const int B = 8;

  const float* Xin = (const float*)d_in[0];
  const float* CLS = (const float*)d_in[1];
  const float* HEAD_B = (const float*)d_in[2];
  const float* HEAD_W = (const float*)d_in[3];
  int sbase[4];
  {
    int b4 = 4;
    for (int i = 0; i < 4; ++i) { sbase[i] = b4; b4 += 12 * DEP[i] + 3; }
  }
  auto P = [&](int i) -> const float* { return (const float*)d_in[i]; };
  enum { AFF1_B = 0, AFF1_W, AFF2_B, AFF2_W, FC1_B, FC1_W, FC2_B, FC2_W,
         KV_W, PROJ_B, PROJ_W, Q_W };

  char* wsp = (char*)d_ws;
  auto alloc = [&](size_t bytes) -> void* {
    void* r = wsp;
    wsp += (bytes + 255) & ~(size_t)255;
    return r;
  };
  const size_t MAXTOK = (size_t)8 * 3136 * 64;
  float* tokA = (float*)alloc(MAXTOK * 4);
  float* tokB = (float*)alloc(MAXTOK * 4);
  float* affF = (float*)alloc(MAXTOK * 4);
  float* qF = (float*)alloc(MAXTOK * 4);
  float* kvF = (float*)alloc(2 * MAXTOK * 4);
  float* hidF = (float*)alloc(4 * MAXTOK * 4);
  float* projF = (float*)alloc(MAXTOK * 4);
  float* attnF = (float*)alloc(MAXTOK * 4);
  float* imcF = (float*)alloc(MAXTOK * 4);
  float* scoF = (float*)alloc((size_t)3136 * 3136 * 4);
  signed char* actq = (signed char*)alloc((size_t)8 * 1024 * 1024);
  signed char* qq = (signed char*)alloc((size_t)2 * 1024 * 1024);
  signed char* kvq = (signed char*)alloc((size_t)4 * 1024 * 1024);
  signed char* probsq = (signed char*)alloc((size_t)3136 * 3136);
  signed char* vtq = (signed char*)alloc((size_t)64 * 3200);
  signed char* wq = (signed char*)alloc((size_t)2 * 1024 * 1024);

  auto RU64 = [](int v) { return (v + 63) & ~63; };
  auto ew = [](long n) { return dim3((unsigned)((n + 255) / 256)); };
  auto qpad = [&](const float* src, long stride, int M, int K, signed char* dst, int Kp) {
    long total = (long)RU64(M) * Kp;
    k_quant_pad<<<ew(total), dim3(256), 0, stream>>>(src, stride, M, K, dst, Kp, total);
  };
  auto gemm = [&](const signed char* A, int lda, const signed char* Bm, int ldb,
                  float* C, int ldc, int M, int N, int Kp, float scale,
                  const float* bias, int relu) {
    dim3 g((N + 63) / 64, (M + 31) / 32);
    k_gemm_i8<<<g, dim3(128), 0, stream>>>(A, lda, Bm, ldb, C, ldc, M, N, Kp, scale,
                                           bias, relu);
  };

  const float DQ = 1.0f / 256.0f;

  float* tok = tokA;
  float* tok2 = tokB;
  int grid = 224;

  for (int s = 0; s < 4; ++s) {
    const int E = EMB[s], p = PS[s], Cin = INC[s], H = NH[s], D = DEP[s];
    const int prevGrid = grid;
    grid /= p;
    int Ntok = grid * grid;
    const int K = Cin * p * p, Kp = RU64(K);
    const long Mrows = (long)B * Ntok;

    {
      long total = Mrows * (long)K;
      k_im2col<<<ew(total), dim3(256), 0, stream>>>(
          (s == 0) ? Xin : tok, imcF, Cin, prevGrid, p, (s != 0) ? 1 : 0, total);
      qpad(imcF, K, (int)Mrows, K, actq, Kp);
      const float* PB = P(sbase[s] + 12 * D + 0);
      const float* PW = P(sbase[s] + 12 * D + 1);
      const float* POS = P(sbase[s] + 12 * D + 2);
      qpad(PW, K, E, K, wq, Kp);
      gemm(actq, Kp, wq, Kp, tok2, E, (int)Mrows, E, Kp, DQ, PB, 0);

      if (s == 3) {
        long tot = (long)B * (Ntok + 1) * E;
        k_concat_cls<<<ew(tot), dim3(256), 0, stream>>>(CLS, tok2, tok, B, Ntok, E, tot);
        Ntok += 1;
        long n = (long)B * Ntok * E;
        k_add_qz<<<ew(n), dim3(256), 0, stream>>>(tok, POS, tok2, n, (long)Ntok * E);
        float* t = tok; tok = tok2; tok2 = t;
      } else {
        long n = (long)B * Ntok * E;
        k_add_qz<<<ew(n), dim3(256), 0, stream>>>(tok2, POS, tok, n, (long)Ntok * E);
      }
    }

    const int NtokP = RU64(Ntok);
    const long nel = (long)B * Ntok * E;
    const float ATTN_SCALE = DQ * 0.125f;

    for (int blk = 0; blk < D; ++blk) {
      const int fb = sbase[s] + 12 * blk;
      const float* AB1 = P(fb + AFF1_B);
      const float* AW1 = P(fb + AFF1_W);
      const float* AB2 = P(fb + AFF2_B);
      const float* AW2 = P(fb + AFF2_W);
      const float* FC1B = P(fb + FC1_B);
      const float* FC1W = P(fb + FC1_W);
      const float* FC2B = P(fb + FC2_B);
      const float* FC2W = P(fb + FC2_W);
      const float* KVW = P(fb + KV_W);
      const float* PRB = P(fb + PROJ_B);
      const float* PRW = P(fb + PROJ_W);
      const float* QW = P(fb + Q_W);

      k_affine<<<ew(nel), dim3(256), 0, stream>>>(tok, affF, nel, AW1, AB1);
      qpad(affF, E, B * Ntok, E, actq, E);
      qpad(QW, E, E, E, wq, E);
      gemm(actq, E, wq, E, qF, E, B * Ntok, E, E, DQ, nullptr, 0);
      qpad(KVW, E, 2 * E, E, wq, E);
      gemm(actq, E, wq, E, kvF, 2 * E, B * Ntok, 2 * E, E, DQ, nullptr, 0);
      qpad(qF, E, B * Ntok, E, qq, E);
      qpad(kvF, 2 * E, B * Ntok, 2 * E, kvq, 2 * E);

      for (int b = 0; b < B; ++b) {
        for (int h = 0; h < H; ++h) {
          const signed char* Aslice = qq + (size_t)b * Ntok * E + h * 64;
          const signed char* Kslice = kvq + (size_t)b * Ntok * (2 * E) + h * 64;
          gemm(Aslice, E, Kslice, 2 * E, scoF, Ntok, Ntok, Ntok, 64, ATTN_SCALE,
               nullptr, 0);
          k_softmax<<<dim3(Ntok), dim3(256), 0, stream>>>(scoF, scoF, Ntok);
          qpad(scoF, Ntok, Ntok, Ntok, probsq, NtokP);
          k_quant_transpose<<<ew((long)64 * NtokP), dim3(256), 0, stream>>>(
              kvF + (size_t)b * Ntok * (2 * E) + E + h * 64, 2 * E, Ntok, NtokP, vtq);
          gemm(probsq, NtokP, vtq, NtokP,
               attnF + (size_t)b * Ntok * E + h * 64, E, Ntok, 64, NtokP, DQ,
               nullptr, 0);
        }
      }
      qpad(attnF, E, B * Ntok, E, actq, E);
      qpad(PRW, E, E, E, wq, E);
      gemm(actq, E, wq, E, projF, E, B * Ntok, E, E, DQ, PRB, 0);
      k_add_qz<<<ew(nel), dim3(256), 0, stream>>>(tok, projF, tok2, nel, nel);
      { float* t = tok; tok = tok2; tok2 = t; }

      k_affine<<<ew(nel), dim3(256), 0, stream>>>(tok, affF, nel, AW2, AB2);
      qpad(affF, E, B * Ntok, E, actq, E);
      qpad(FC1W, E, 4 * E, E, wq, E);
      gemm(actq, E, wq, E, hidF, 4 * E, B * Ntok, 4 * E, E, DQ, FC1B, 1);
      qpad(hidF, 4 * E, B * Ntok, 4 * E, actq, 4 * E);
      qpad(FC2W, 4 * E, E, 4 * E, wq, 4 * E);
      gemm(actq, 4 * E, wq, 4 * E, projF, E, B * Ntok, E, 4 * E, DQ, FC2B, 0);
      k_add_qz<<<ew(nel), dim3(256), 0, stream>>>(tok, projF, tok2, nel, nel);
      { float* t = tok; tok = tok2; tok2 = t; }
    }
  }

  {
    const int E4 = 512, NC = 1000, NTOK4 = 50;
    qpad(tok, (long)NTOK4 * E4, B, E4, actq, E4);
    qpad(HEAD_W, E4, NC, E4, wq, E4);
    gemm(actq, E4, wq, E4, (float*)d_out, NC, B, NC, E4, DQ, HEAD_B, 0);
  }
}
